// Qwen3MoeExperts_32049045963285
// MI455X (gfx1250) — matrix operand order repaired and checked by reading, not compiled, not run on hardware
//
#include <hip/hip_runtime.h>
#include <hip/hip_bf16.h>

// Problem constants (from reference): E=8, H=1024, I=768, K=2, T=2048
#define T_TOK 2048
#define HDIM  1024
#define IDIM  768
#define NEXP  8
#define TOPK  2
// Worst-case padded tiles: sum_e ceil(cnt_e/16) <= 256 + 8*... <= 264
#define MAXTILES 264
#define MAXROWS  (MAXTILES * 16)

typedef __attribute__((ext_vector_type(16))) _Float16 v16h;
typedef __attribute__((ext_vector_type(8)))  _Float16 v8h;
typedef __attribute__((ext_vector_type(8)))  float    v8f;
typedef __attribute__((ext_vector_type(4)))  float    f32x4;
typedef __attribute__((ext_vector_type(4)))  _Float16 f16x4;

__device__ __forceinline__ v16h cat8(v8h lo, v8h hi) {
    return __builtin_shufflevector(lo, hi, 0, 1, 2, 3, 4, 5, 6, 7,
                                           8, 9, 10, 11, 12, 13, 14, 15);
}

// ---------------------------------------------------------------------------
// 1) f32 -> f16 weight conversion (grid-stride, float4 vectorized)
// ---------------------------------------------------------------------------
__global__ void cvt_f32_to_f16(const float* __restrict__ src,
                               _Float16* __restrict__ dst, int n4) {
    int stride = gridDim.x * blockDim.x;
    for (int i = blockIdx.x * blockDim.x + threadIdx.x; i < n4; i += stride) {
        f32x4 v = ((const f32x4*)src)[i];
        f16x4 o;
        o[0] = (_Float16)v[0];
        o[1] = (_Float16)v[1];
        o[2] = (_Float16)v[2];
        o[3] = (_Float16)v[3];
        ((f16x4*)dst)[i] = o;
    }
}

// ---------------------------------------------------------------------------
// 2) Routing: per-expert counts, 16-row-padded segments, slot assignment
//    (single block of 256 threads)
// ---------------------------------------------------------------------------
__global__ void route_kernel(const int* __restrict__ topk,
                             int* __restrict__ rowtoken,     // [MAXROWS]
                             int* __restrict__ row_of,       // [T*K]
                             int* __restrict__ tile_expert)  // [MAXTILES]
{
    __shared__ int cnt[NEXP];
    __shared__ int cur[NEXP];
    const int tid = threadIdx.x;

    if (tid < NEXP) cnt[tid] = 0;
    for (int i = tid; i < MAXROWS; i += 256) rowtoken[i] = -1;
    for (int i = tid; i < MAXTILES; i += 256) tile_expert[i] = -1;
    __syncthreads();

    for (int i = tid; i < T_TOK * TOPK; i += 256) {
        int e = topk[i];
        atomicAdd(&cnt[e], 1);
    }
    __syncthreads();

    if (tid == 0) {
        int o = 0, tile = 0;
        for (int e = 0; e < NEXP; ++e) {
            cur[e] = o;
            int tiles = (cnt[e] + 15) >> 4;
            for (int j = 0; j < tiles; ++j) tile_expert[tile++] = e;
            o += tiles * 16;
        }
    }
    __syncthreads();

    for (int i = tid; i < T_TOK * TOPK; i += 256) {
        int e = topk[i];
        int pos = atomicAdd(&cur[e], 1);
        rowtoken[pos] = i / TOPK;
        row_of[i] = pos;
    }
}

// ---------------------------------------------------------------------------
// 3) Gather hidden_states rows -> f16 A buffer (one block per padded row)
// ---------------------------------------------------------------------------
__global__ void gather_kernel(const float* __restrict__ hs,
                              const int* __restrict__ rowtoken,
                              _Float16* __restrict__ A) {
    const int row = blockIdx.x;
    const int tid = threadIdx.x;  // 256 threads, 4 floats each = 1024
    const int t = rowtoken[row];
    f16x4 o;
    if (t >= 0) {
        f32x4 v = ((const f32x4*)(hs + (size_t)t * HDIM))[tid];
        o[0] = (_Float16)v[0];
        o[1] = (_Float16)v[1];
        o[2] = (_Float16)v[2];
        o[3] = (_Float16)v[3];
    } else {
        o[0] = (_Float16)0.f; o[1] = (_Float16)0.f;
        o[2] = (_Float16)0.f; o[3] = (_Float16)0.f;
    }
    ((f16x4*)(A + (size_t)row * HDIM))[tid] = o;
}

// ---------------------------------------------------------------------------
// 4) Grouped GEMM: one block (8 wave32) per 16-row expert tile.
//    Stage1: gu = A @ Wgu^T, h = silu(gate)*up -> LDS (f16)
//    Stage2: Hout = h @ Wd^T (f32 rows to workspace)
// ---------------------------------------------------------------------------
__global__ __launch_bounds__(256) void moe_gemm_kernel(
    const _Float16* __restrict__ A,    // [MAXROWS, H]
    const _Float16* __restrict__ Wg,   // [E, 2I, H]
    const _Float16* __restrict__ Wd,   // [E, H, I]
    const int* __restrict__ tile_expert,
    float* __restrict__ Hout)          // [MAXROWS, H]
{
    __shared__ __align__(16) _Float16 hbuf[16 * IDIM];  // 24 KB

    const int tile = blockIdx.x;
    const int e = tile_expert[tile];
    if (e < 0) return;

    const int tid  = threadIdx.x;
    const int wave = tid >> 5;
    const int lane = tid & 31;
    const int half = lane >> 4;   // which 16-lane half
    const int l15  = lane & 15;   // row (A) / column (B/D) index within tile

    // ---- Stage 1: gate/up projection + silu fusion -------------------------
    {
        // A-fragment row for this lane (contiguous K layout:
        // lanes 0-15 take K kb..kb+15, lanes 16-31 take K kb+16..kb+31)
        const _Float16* Arow = A + (size_t)(tile * 16 + l15) * HDIM + half * 16;
        const _Float16* We   = Wg + (size_t)e * (2 * IDIM) * HDIM;

        for (int nt = wave; nt < IDIM / 16; nt += 8) {
            v8f accG = {0.f, 0.f, 0.f, 0.f, 0.f, 0.f, 0.f, 0.f};
            v8f accU = {0.f, 0.f, 0.f, 0.f, 0.f, 0.f, 0.f, 0.f};
            // B layout: lane holds a contiguous 16-K run for its column;
            // lanes 16-31 take K base + 16.
            const _Float16* Bg = We + (size_t)(nt * 16 + l15) * HDIM + half * 16;
            const _Float16* Bu = We + (size_t)(IDIM + nt * 16 + l15) * HDIM + half * 16;

#pragma unroll 4
            for (int kb = 0; kb < HDIM; kb += 32) {
                v16h a = cat8(*(const v8h*)(Arow + kb),
                              *(const v8h*)(Arow + kb + 8));
                v16h bg = cat8(*(const v8h*)(Bg + kb),
                               *(const v8h*)(Bg + kb + 8));
                v16h bu = cat8(*(const v8h*)(Bu + kb),
                               *(const v8h*)(Bu + kb + 8));
                accG = __builtin_amdgcn_wmma_f32_16x16x32_f16(
                    false, a, false, bg, (short)0, accG, false, false);
                accU = __builtin_amdgcn_wmma_f32_16x16x32_f16(
                    false, a, false, bu, (short)0, accU, false, false);
            }
            // D layout: element i -> row M = i + 8*half, col N = l15
#pragma unroll
            for (int i = 0; i < 8; ++i) {
                float g = accG[i];
                float u = accU[i];
                float hv = (g / (1.0f + __expf(-g))) * u;  // silu(g)*u
                int m = i + half * 8;
                hbuf[m * IDIM + nt * 16 + l15] = (_Float16)hv;
            }
        }
    }
    __syncthreads();

    // ---- Stage 2: down projection -----------------------------------------
    {
        const _Float16* De   = Wd + (size_t)e * HDIM * IDIM;
        const _Float16* hrow = hbuf + l15 * IDIM + half * 16;

        for (int nt = wave; nt < HDIM / 16; nt += 8) {
            v8f acc = {0.f, 0.f, 0.f, 0.f, 0.f, 0.f, 0.f, 0.f};
            const _Float16* Brow = De + (size_t)(nt * 16 + l15) * IDIM + half * 16;

#pragma unroll 4
            for (int kb = 0; kb < IDIM; kb += 32) {
                v16h a = cat8(*(const v8h*)(hrow + kb),
                              *(const v8h*)(hrow + kb + 8));
                v16h b = cat8(*(const v8h*)(Brow + kb),
                              *(const v8h*)(Brow + kb + 8));
                acc = __builtin_amdgcn_wmma_f32_16x16x32_f16(
                    false, a, false, b, (short)0, acc, false, false);
            }
#pragma unroll
            for (int i = 0; i < 8; ++i) {
                int m = i + half * 8;
                Hout[(size_t)(tile * 16 + m) * HDIM + nt * 16 + l15] = acc[i];
            }
        }
    }
}

// ---------------------------------------------------------------------------
// 5) Combine: out[t] = sum_k w[t,k] * Hout[row_of[t,k]]  (no atomics)
// ---------------------------------------------------------------------------
__global__ void combine_kernel(const float* __restrict__ wts,
                               const int* __restrict__ row_of,
                               const float* __restrict__ Hout,
                               float* __restrict__ out) {
    const int t = blockIdx.x;
    const int r0 = row_of[t * TOPK + 0];
    const int r1 = row_of[t * TOPK + 1];
    const float w0 = wts[t * TOPK + 0];
    const float w1 = wts[t * TOPK + 1];
    const f32x4* h0 = (const f32x4*)(Hout + (size_t)r0 * HDIM);
    const f32x4* h1 = (const f32x4*)(Hout + (size_t)r1 * HDIM);
    f32x4* o = (f32x4*)(out + (size_t)t * HDIM);
    const int i = threadIdx.x;  // 256 threads * 4 floats = 1024
    f32x4 a = h0[i], b = h1[i], r;
    r[0] = w0 * a[0] + w1 * b[0];
    r[1] = w0 * a[1] + w1 * b[1];
    r[2] = w0 * a[2] + w1 * b[2];
    r[3] = w0 * a[3] + w1 * b[3];
    o[i] = r;
}

// ---------------------------------------------------------------------------
extern "C" void kernel_launch(void* const* d_in, const int* in_sizes, int n_in,
                              void* d_out, int out_size, void* d_ws, size_t ws_size,
                              hipStream_t stream) {
    (void)in_sizes; (void)n_in; (void)out_size; (void)ws_size;
    const float* hs   = (const float*)d_in[0];  // [T, H]
    const float* gup  = (const float*)d_in[1];  // [E, 2I, H]
    const float* dwn  = (const float*)d_in[2];  // [E, H, I]
    const int*   topk = (const int*)d_in[3];    // [T, K]
    const float* wts  = (const float*)d_in[4];  // [T, K]
    float* out = (float*)d_out;                 // [T, H]

    // Workspace carve-out (256B aligned slices)
    char* ws = (char*)d_ws;
    size_t off = 0;
    auto carve = [&](size_t bytes) -> void* {
        void* p = ws + off;
        off = (off + bytes + 255) & ~(size_t)255;
        return p;
    };
    const size_t n_gu = (size_t)NEXP * 2 * IDIM * HDIM;  // 12.58M
    const size_t n_dn = (size_t)NEXP * HDIM * IDIM;      // 6.29M
    _Float16* Wg   = (_Float16*)carve(n_gu * sizeof(_Float16));        // 25.2 MB
    _Float16* Wd   = (_Float16*)carve(n_dn * sizeof(_Float16));        // 12.6 MB
    _Float16* Abuf = (_Float16*)carve((size_t)MAXROWS * HDIM * 2);     //  8.7 MB
    float*    Hout = (float*)carve((size_t)MAXROWS * HDIM * 4);        // 17.3 MB
    int* rowtoken    = (int*)carve(MAXROWS * sizeof(int));
    int* row_of      = (int*)carve((size_t)T_TOK * TOPK * sizeof(int));
    int* tile_expert = (int*)carve(MAXTILES * sizeof(int));

    // 1) weights -> f16 (amortized across all tiles of each expert)
    cvt_f32_to_f16<<<2048, 256, 0, stream>>>(gup, Wg, (int)(n_gu / 4));
    cvt_f32_to_f16<<<2048, 256, 0, stream>>>(dwn, Wd, (int)(n_dn / 4));
    // 2) routing / slot assignment (single block)
    route_kernel<<<1, 256, 0, stream>>>(topk, rowtoken, row_of, tile_expert);
    // 3) gather tokens into per-expert padded row blocks
    gather_kernel<<<MAXROWS, 256, 0, stream>>>(hs, rowtoken, Abuf);
    // 4) grouped GEMM: gate/up + silu + down, WMMA f16->f32
    moe_gemm_kernel<<<MAXTILES, 256, 0, stream>>>(Abuf, Wg, Wd, tile_expert, Hout);
    // 5) weighted combine back to token order
    combine_kernel<<<T_TOK, 256, 0, stream>>>(wts, row_of, Hout, out);
}
